// EGNNLayer_86019605004410
// MI455X (gfx1250) — matrix operand order repaired and checked by reading, not compiled, not run on hardware
//
#include <hip/hip_runtime.h>
#include <stdint.h>

// ---------------------------------------------------------------------------
// EGNN layer for MI455X (gfx1250, wave32, WMMA).
// Edge MLPs dominate (~58 GFLOP vs ~90MB traffic -> compute bound): all GEMMs
// as bf16 x bf16 -> f32 WMMA (v_wmma_f32_16x16x32_bf16) on 16-row tiles.
// Weights pre-packed into the ISA 32x16 B-matrix lane layout, staged into LDS
// with GLOBAL_LOAD_ASYNC_TO_LDS_B128 (ASYNCcnt). Dual accumulators per GEMM
// break the WMMA->WMMA RAW chain; silu epilogue uses v_rcp_f32/v_exp_f32
// (TRANS pipe) instead of the precise-division macro sequence.
// ---------------------------------------------------------------------------

#define H_DIM   128
#define ED_DIM  48
#define KMSG    305          // 2H + ED + 1
#define KMSG_P  320          // padded to 10 K-chunks of 32
#define LN_EPS  1e-5f

typedef __attribute__((ext_vector_type(16))) __bf16 v16bf;
typedef __attribute__((ext_vector_type(8)))  float  v8f;

#define WMMA_BF16(a, b, c) \
  __builtin_amdgcn_wmma_f32_16x16x32_bf16(false, (a), false, (b), (short)0, (c), false, false)

// ------------------------------ helpers ------------------------------------

__device__ __forceinline__ void ld8(const float* __restrict__ p, float* r) {
  const float4* q = (const float4*)p;
  float4 a = q[0], b = q[1];
  r[0]=a.x; r[1]=a.y; r[2]=a.z; r[3]=a.w;
  r[4]=b.x; r[5]=b.y; r[6]=b.z; r[7]=b.w;
}

// msg_input row element layout: [h[src](0..127) | h[dst](128..255) |
//                                edge_attr(256..303) | d2(304) | pad..319]
// k0 is always a multiple of 8, and every 8-run stays inside one region.
__device__ __forceinline__ void load_msg8(const float* __restrict__ h,
                                          const float* __restrict__ ea,
                                          long long s, long long d, long long e,
                                          float d2v, int k0, float* r) {
  if (k0 < 128) {
    ld8(h + s * H_DIM + k0, r);
  } else if (k0 < 256) {
    ld8(h + d * H_DIM + (k0 - 128), r);
  } else if (k0 < 304) {
    ld8(ea + e * ED_DIM + (k0 - 256), r);
  } else if (k0 == 304) {
    r[0] = d2v;
    #pragma unroll
    for (int j = 1; j < 8; ++j) r[j] = 0.f;
  } else {
    #pragma unroll
    for (int j = 0; j < 8; ++j) r[j] = 0.f;
  }
}

// node MLP input row: [h(0..127) | agg(128..255)]
__device__ __forceinline__ void load_node8(const float* __restrict__ h,
                                           const float* __restrict__ agg,
                                           long long n, int k0, float* r) {
  if (k0 < 128) ld8(h + n * H_DIM + k0, r);
  else          ld8(agg + n * H_DIM + (k0 - 128), r);
}

__device__ __forceinline__ v16bf pack_a(const float* r0, const float* r1) {
  v16bf a;
  #pragma unroll
  for (int j = 0; j < 8; ++j) {
    a[j]     = (__bf16)r0[j];
    a[8 + j] = (__bf16)r1[j];
  }
  return a;
}

// B tile: 1024 bytes per (kchunk,ntile), lane-major 32 bytes (16 bf16) per lane.
__device__ __forceinline__ v16bf lds_btile(const char* smem, int baseOff,
                                           int c, int t, int lane) {
  return *(const v16bf*)(smem + baseOff + ((c * 8 + t) << 10) + (lane << 5));
}

// silu on the TRANS pipe: v_exp_f32 + v_rcp_f32 (result feeds bf16 WMMA input
// or a clipped scalar, so 1-ulp rcp is ample precision).
__device__ __forceinline__ float silu(float v) {
  return v * __builtin_amdgcn_rcpf(1.f + __expf(-v));
}

// Cooperative global->LDS stage on the CDNA5 async path.
// Low 32 bits of a generic pointer into __shared__ are the allocation-relative
// LDS address (ISA 10.2), which is what the async-load
// dsaddr VGPR expects. Each wave fences its own ASYNCcnt before the barrier.
__device__ __forceinline__ void async_stage(const char* __restrict__ gsrc,
                                            char* ldst, int bytes) {
  const unsigned lds0 = (unsigned)(uintptr_t)ldst;
  for (int i = threadIdx.x * 16; i < bytes; i += blockDim.x * 16) {
    unsigned long long ga = (unsigned long long)(uintptr_t)(gsrc + i);
    unsigned la = lds0 + (unsigned)i;
    asm volatile("global_load_async_to_lds_b128 %0, %1, off"
                 :: "v"(la), "v"(ga) : "memory");
  }
  asm volatile("s_wait_asynccnt 0x0" ::: "memory");
}

// ------------------------- weight repack (f32 -> bf16 B layout) -------------
// B-matrix 32x16 bf16 layout (ISA 7.12.2): lanes 0-15 hold K=0..7,16..23,
// lanes 16-31 hold K=8..15,24..31, column N = lane&15, 16 packed K per lane.
__global__ void pack_w_kernel(const float* __restrict__ W, __bf16* __restrict__ out,
                              int K, int KP) {
  int total = (KP / 32) * 8 * 512;            // 512 bf16 per 32x16 tile
  for (int idx = blockIdx.x * blockDim.x + threadIdx.x; idx < total;
       idx += gridDim.x * blockDim.x) {
    int tile   = idx >> 9;
    int within = idx & 511;
    int lane = within >> 4;
    int j    = within & 15;
    int c = tile >> 3;
    int t = tile & 7;
    int k = c * 32 + ((lane >> 4) << 3) + (j & 7) + ((j >> 3) << 4);
    int n = t * 16 + (lane & 15);
    float v = (k < K) ? W[k * H_DIM + n] : 0.f;
    out[idx] = (__bf16)v;
  }
}

__global__ void zero_kernel(float* __restrict__ p, long long n) {
  for (long long i = blockIdx.x * (long long)blockDim.x + threadIdx.x; i < n;
       i += (long long)gridDim.x * blockDim.x)
    p[i] = 0.f;
}

// ------------------------------- edge kernel --------------------------------
// 256 threads = 8 waves; each wave owns one 16-edge tile per iteration.
// LDS: Wm1p 80K | Wm2p 32K | Wc1p 32K | biases | 8x per-wave scratch.
#define E_WM1_OFF   0
#define E_WM2_OFF   81920
#define E_WC1_OFF   114688
#define E_BIAS_OFF  147456          // bm1|bm2|bc1|wc2 (4x128 f32) + bc2
#define E_SCR_OFF   149760
#define E_SCR_STR   8256            // 16x128 f32 msg scratch + 16 dst ints
#define E_SMEM_SZ   (E_SCR_OFF + 8 * E_SCR_STR)   // 215808 B < 320KB WGP LDS

__global__ void __launch_bounds__(256, 1)
edge_kernel(const float* __restrict__ x, const float* __restrict__ h,
            const long long* __restrict__ eidx, const float* __restrict__ eattr,
            const __bf16* __restrict__ Wm1p, const float* __restrict__ bm1,
            const float* __restrict__ bm2, const float* __restrict__ bc1,
            const float* __restrict__ Wc2, const float* __restrict__ bc2,
            float* __restrict__ agg, float* __restrict__ coord, int E) {
  extern __shared__ __align__(16) char smem[];

  // stage packed Wm1|Wm2|Wc1 (contiguous in ws, 147456 B) via async-to-LDS
  async_stage((const char*)Wm1p, smem, 147456);
  {
    float* bL = (float*)(smem + E_BIAS_OFF);
    for (int i = threadIdx.x; i < 513; i += blockDim.x) {
      float v;
      if (i < 128)      v = bm1[i];
      else if (i < 256) v = bm2[i - 128];
      else if (i < 384) v = bc1[i - 256];
      else if (i < 512) v = Wc2[i - 384];
      else              v = bc2[0];
      bL[i] = v;
    }
  }
  __syncthreads();

  const int lane = threadIdx.x & 31;
  const int wave = threadIdx.x >> 5;
  const int hi   = lane >> 4;          // which K-half of the A layout
  const int rowm = lane & 15;          // matrix row (edge within tile)

  const float* bL  = (const float*)(smem + E_BIAS_OFF);
  const float* bm1L = bL;
  const float* bm2L = bL + 128;
  const float* bc1L = bL + 256;
  const float* wc2L = bL + 384;
  const float  bc2v = bL[512];

  float* scratch = (float*)(smem + E_SCR_OFF + wave * E_SCR_STR);
  int*   dstLds  = (int*)(scratch + 16 * 128);

  const int ntiles = E >> 4;           // E is a multiple of 16 for this problem
  const int stride = gridDim.x * 8;
  for (int tile = blockIdx.x * 8 + wave; tile < ntiles; tile += stride) {
    const long long e = (long long)tile * 16 + rowm;
    const long long s = eidx[e];
    const long long d = eidx[(long long)E + e];
    if (hi == 0) dstLds[rowm] = (int)d;

    // prefetch the only streamed (non-L2-resident) operands of the next tile
    const int nxt = tile + stride;
    if (nxt < ntiles) {
      __builtin_prefetch(eattr + ((long long)nxt * 16 + rowm) * ED_DIM, 0, 1);
      __builtin_prefetch(eidx + (long long)nxt * 16, 0, 1);
    }

    const float rx = x[s * 3 + 0] - x[d * 3 + 0];
    const float ry = x[s * 3 + 1] - x[d * 3 + 1];
    const float rz = x[s * 3 + 2] - x[d * 3 + 2];
    const float d2 = rx * rx + ry * ry + rz * rz;

    // ---- build A (16x320 bf16, ISA 16-bit A layout) --------------------
    v16bf A[10];
    #pragma unroll
    for (int c = 0; c < 10; ++c) {
      const int k0 = c * 32 + hi * 8;
      float r0[8], r1[8];
      load_msg8(h, eattr, s, d, e, d2, k0, r0);
      load_msg8(h, eattr, s, d, e, d2, k0 + 16, r1);
      A[c] = pack_a(r0, r1);
    }

    // ---- GEMM1 (16x320 x 320x128) + bm1 + silu -> scratch --------------
    #pragma unroll 1
    for (int t = 0; t < 8; ++t) {
      v8f acc0 = {}, acc1 = {};        // dual accumulators: break RAW chain
      #pragma unroll
      for (int c = 0; c < 10; c += 2) {
        acc0 = WMMA_BF16(A[c],     lds_btile(smem, E_WM1_OFF, c,     t, lane), acc0);
        acc1 = WMMA_BF16(A[c + 1], lds_btile(smem, E_WM1_OFF, c + 1, t, lane), acc1);
      }
      const int col = t * 16 + rowm;
      const float bias = bm1L[col];
      #pragma unroll
      for (int r = 0; r < 8; ++r) {
        const int m = r + 8 * hi;
        scratch[m * 128 + col] = silu(acc0[r] + acc1[r] + bias);
      }
    }

    // ---- A2 <- scratch -------------------------------------------------
    v16bf A2[4];
    #pragma unroll
    for (int c = 0; c < 4; ++c) {
      const int k0 = c * 32 + hi * 8;
      float r0[8], r1[8];
      #pragma unroll
      for (int j = 0; j < 8; ++j) {
        r0[j] = scratch[rowm * 128 + k0 + j];
        r1[j] = scratch[rowm * 128 + k0 + 16 + j];
      }
      A2[c] = pack_a(r0, r1);
    }

    // ---- GEMM2 (msg = silu1 x Wm2 + bm2) -> scratch --------------------
    #pragma unroll 1
    for (int t = 0; t < 8; ++t) {
      v8f acc0 = {}, acc1 = {};
      acc0 = WMMA_BF16(A2[0], lds_btile(smem, E_WM2_OFF, 0, t, lane), acc0);
      acc1 = WMMA_BF16(A2[1], lds_btile(smem, E_WM2_OFF, 1, t, lane), acc1);
      acc0 = WMMA_BF16(A2[2], lds_btile(smem, E_WM2_OFF, 2, t, lane), acc0);
      acc1 = WMMA_BF16(A2[3], lds_btile(smem, E_WM2_OFF, 3, t, lane), acc1);
      const int col = t * 16 + rowm;
      const float bias = bm2L[col];
      #pragma unroll
      for (int r = 0; r < 8; ++r) {
        const int m = r + 8 * hi;
        scratch[m * 128 + col] = acc0[r] + acc1[r] + bias;
      }
    }

    // ---- A3 <- msg (for coord-weight MLP) ------------------------------
    v16bf A3[4];
    #pragma unroll
    for (int c = 0; c < 4; ++c) {
      const int k0 = c * 32 + hi * 8;
      float r0[8], r1[8];
      #pragma unroll
      for (int j = 0; j < 8; ++j) {
        r0[j] = scratch[rowm * 128 + k0 + j];
        r1[j] = scratch[rowm * 128 + k0 + 16 + j];
      }
      A3[c] = pack_a(r0, r1);
    }

    // ---- scatter msg into agg (segment_sum via f32 atomics) ------------
    // flat = lane + 32*i: each instruction covers 32 contiguous columns of
    // one agg row -> coalesces to a single 128B L2 segment.
    #pragma unroll 1
    for (int i = 0; i < 64; ++i) {
      const int flat = lane + i * 32;
      const int m = flat >> 7;
      const int c = flat & 127;
      atomicAdd(&agg[(long long)dstLds[m] * H_DIM + c], scratch[flat]);
    }

    // ---- GEMM3 (silu(msg x Wc1 + bc1)) -> scratch ----------------------
    #pragma unroll 1
    for (int t = 0; t < 8; ++t) {
      v8f acc0 = {}, acc1 = {};
      acc0 = WMMA_BF16(A3[0], lds_btile(smem, E_WC1_OFF, 0, t, lane), acc0);
      acc1 = WMMA_BF16(A3[1], lds_btile(smem, E_WC1_OFF, 1, t, lane), acc1);
      acc0 = WMMA_BF16(A3[2], lds_btile(smem, E_WC1_OFF, 2, t, lane), acc0);
      acc1 = WMMA_BF16(A3[3], lds_btile(smem, E_WC1_OFF, 3, t, lane), acc1);
      const int col = t * 16 + rowm;
      const float bias = bc1L[col];
      #pragma unroll
      for (int r = 0; r < 8; ++r) {
        const int m = r + 8 * hi;
        scratch[m * 128 + col] = silu(acc0[r] + acc1[r] + bias);
      }
    }

    // ---- w = clip(. x Wc2 + bc2); coord scatter ------------------------
    if (lane < 16) {
      float sum = bc2v;
      #pragma unroll 4
      for (int c = 0; c < 128; ++c) sum += scratch[lane * 128 + c] * wc2L[c];
      sum = fminf(1.f, fmaxf(-1.f, sum));
      atomicAdd(&coord[d * 3 + 0], sum * rx);
      atomicAdd(&coord[d * 3 + 1], sum * ry);
      atomicAdd(&coord[d * 3 + 2], sum * rz);
    }
  }
}

// ------------------------------- node kernel --------------------------------
#define N_WN1_OFF   0
#define N_WN2_OFF   65536
#define N_BIAS_OFF  98304           // bn1|bn2|gamma|beta (4x128 f32)
#define N_SCR_OFF   100352
#define N_SCR_STR   8192
#define N_SMEM_SZ   (N_SCR_OFF + 8 * N_SCR_STR)   // 165888 B

__global__ void __launch_bounds__(256, 1)
node_kernel(const float* __restrict__ x, const float* __restrict__ h,
            const float* __restrict__ agg, const float* __restrict__ coord,
            const __bf16* __restrict__ Wn1p,
            const float* __restrict__ bn1, const float* __restrict__ bn2,
            const float* __restrict__ gamma, const float* __restrict__ beta,
            float* __restrict__ out, int N) {
  extern __shared__ __align__(16) char smem[];
  async_stage((const char*)Wn1p, smem, 98304);   // Wn1p|Wn2p contiguous
  {
    float* bL = (float*)(smem + N_BIAS_OFF);
    for (int i = threadIdx.x; i < 512; i += blockDim.x) {
      float v;
      if (i < 128)      v = bn1[i];
      else if (i < 256) v = bn2[i - 128];
      else if (i < 384) v = gamma[i - 256];
      else              v = beta[i - 384];
      bL[i] = v;
    }
  }
  __syncthreads();

  const int lane = threadIdx.x & 31;
  const int wave = threadIdx.x >> 5;
  const int hi   = lane >> 4;
  const int rowm = lane & 15;

  const float* bL   = (const float*)(smem + N_BIAS_OFF);
  const float* bn1L = bL;
  const float* bn2L = bL + 128;
  const float* gmL  = bL + 256;
  const float* btL  = bL + 384;
  float* scratch = (float*)(smem + N_SCR_OFF + wave * N_SCR_STR);
  float* hOut = out + (long long)3 * N;

  const int ntiles = (N + 15) >> 4;
  for (int tile = blockIdx.x * 8 + wave; tile < ntiles; tile += gridDim.x * 8) {
    const long long nr = (long long)tile * 16 + rowm;
    const long long n  = nr < N ? nr : (long long)N - 1;

    v16bf A[8];
    #pragma unroll
    for (int c = 0; c < 8; ++c) {
      const int k0 = c * 32 + hi * 8;
      float r0[8], r1[8];
      load_node8(h, agg, n, k0, r0);
      load_node8(h, agg, n, k0 + 16, r1);
      A[c] = pack_a(r0, r1);
    }

    // GEMM1 ([h|agg] x Wn1) + bn1 + silu -> scratch
    #pragma unroll 1
    for (int t = 0; t < 8; ++t) {
      v8f acc0 = {}, acc1 = {};
      #pragma unroll
      for (int c = 0; c < 8; c += 2) {
        acc0 = WMMA_BF16(A[c],     lds_btile(smem, N_WN1_OFF, c,     t, lane), acc0);
        acc1 = WMMA_BF16(A[c + 1], lds_btile(smem, N_WN1_OFF, c + 1, t, lane), acc1);
      }
      const int col = t * 16 + rowm;
      const float bias = bn1L[col];
      #pragma unroll
      for (int r = 0; r < 8; ++r) {
        const int m = r + 8 * hi;
        scratch[m * 128 + col] = silu(acc0[r] + acc1[r] + bias);
      }
    }

    v16bf A2[4];
    #pragma unroll
    for (int c = 0; c < 4; ++c) {
      const int k0 = c * 32 + hi * 8;
      float r0[8], r1[8];
      #pragma unroll
      for (int j = 0; j < 8; ++j) {
        r0[j] = scratch[rowm * 128 + k0 + j];
        r1[j] = scratch[rowm * 128 + k0 + 16 + j];
      }
      A2[c] = pack_a(r0, r1);
    }

    // GEMM2 + bn2 + residual h -> scratch
    #pragma unroll 1
    for (int t = 0; t < 8; ++t) {
      v8f acc0 = {}, acc1 = {};
      acc0 = WMMA_BF16(A2[0], lds_btile(smem, N_WN2_OFF, 0, t, lane), acc0);
      acc1 = WMMA_BF16(A2[1], lds_btile(smem, N_WN2_OFF, 1, t, lane), acc1);
      acc0 = WMMA_BF16(A2[2], lds_btile(smem, N_WN2_OFF, 2, t, lane), acc0);
      acc1 = WMMA_BF16(A2[3], lds_btile(smem, N_WN2_OFF, 3, t, lane), acc1);
      const int col = t * 16 + rowm;
      const float bias = bn2L[col];
      #pragma unroll
      for (int r = 0; r < 8; ++r) {
        const int m = r + 8 * hi;
        const long long nmr = (long long)tile * 16 + m;
        const long long nm  = nmr < N ? nmr : (long long)N - 1;
        scratch[m * 128 + col] = acc0[r] + acc1[r] + bias + h[nm * H_DIM + col];
      }
    }

    // layernorm + writeback; plus x_new = x + coord_delta
    const long long n2 = (long long)tile * 16 + lane;
    if (lane < 16 && n2 < N) {
      float mu = 0.f;
      #pragma unroll 4
      for (int c = 0; c < 128; ++c) mu += scratch[lane * 128 + c];
      mu *= (1.f / 128.f);
      float var = 0.f;
      #pragma unroll 4
      for (int c = 0; c < 128; ++c) {
        const float dv = scratch[lane * 128 + c] - mu;
        var += dv * dv;
      }
      var *= (1.f / 128.f);
      const float inv = rsqrtf(var + LN_EPS);
      #pragma unroll 4
      for (int c = 0; c < 128; ++c) {
        hOut[n2 * H_DIM + c] =
            (scratch[lane * 128 + c] - mu) * inv * gmL[c] + btL[c];
      }
      out[n2 * 3 + 0] = x[n2 * 3 + 0] + coord[n2 * 3 + 0];
      out[n2 * 3 + 1] = x[n2 * 3 + 1] + coord[n2 * 3 + 1];
      out[n2 * 3 + 2] = x[n2 * 3 + 2] + coord[n2 * 3 + 2];
    }
  }
}

// ------------------------------ launcher ------------------------------------
extern "C" void kernel_launch(void* const* d_in, const int* in_sizes, int n_in,
                              void* d_out, int out_size, void* d_ws, size_t ws_size,
                              hipStream_t stream) {
  const float*     x     = (const float*)d_in[0];
  const float*     h     = (const float*)d_in[1];
  const long long* eidx  = (const long long*)d_in[2];   // int64 per reference
  const float*     eattr = (const float*)d_in[3];
  const float* Wm1 = (const float*)d_in[4];
  const float* bm1 = (const float*)d_in[5];
  const float* Wm2 = (const float*)d_in[6];
  const float* bm2 = (const float*)d_in[7];
  const float* Wn1 = (const float*)d_in[8];
  const float* bn1 = (const float*)d_in[9];
  const float* Wn2 = (const float*)d_in[10];
  const float* bn2 = (const float*)d_in[11];
  const float* Wc1 = (const float*)d_in[12];
  const float* bc1 = (const float*)d_in[13];
  const float* Wc2 = (const float*)d_in[14];
  const float* bc2 = (const float*)d_in[15];
  const float* gamma = (const float*)d_in[16];
  const float* beta  = (const float*)d_in[17];

  const int N = in_sizes[0] / 3;
  const int E = in_sizes[3] / ED_DIM;
  float* out = (float*)d_out;

  // workspace: agg[N*128] | coord[N*3] | packed bf16 weights
  char* ws = (char*)d_ws;
  float* agg   = (float*)ws;
  float* coord = agg + (size_t)N * H_DIM;
  size_t woff = (((size_t)N * 131 * sizeof(float)) + 255) & ~(size_t)255;
  __bf16* Wm1p = (__bf16*)(ws + woff);          // 10 chunks * 8 tiles * 512
  __bf16* Wm2p = Wm1p + 40960;                  //  4 * 8 * 512
  __bf16* Wc1p = Wm2p + 16384;
  __bf16* Wn1p = Wc1p + 16384;                  //  8 * 8 * 512
  __bf16* Wn2p = Wn1p + 32768;
  (void)n_in; (void)out_size; (void)ws_size; (void)Wn2p;

  // zero the segment-sum accumulators (agg and coord are contiguous)
  zero_kernel<<<512, 256, 0, stream>>>(agg, (long long)N * 131);

  // repack weights to bf16 WMMA B-tile layout
  pack_w_kernel<<<160, 256, 0, stream>>>(Wm1, Wm1p, KMSG, KMSG_P);
  pack_w_kernel<<<64,  256, 0, stream>>>(Wm2, Wm2p, H_DIM, H_DIM);
  pack_w_kernel<<<64,  256, 0, stream>>>(Wc1, Wc1p, H_DIM, H_DIM);
  pack_w_kernel<<<128, 256, 0, stream>>>(Wn1, Wn1p, 2 * H_DIM, 2 * H_DIM);
  pack_w_kernel<<<64,  256, 0, stream>>>(Wn2, Wn2p, H_DIM, H_DIM);

  // edge pipeline: 25000 tiles of 16 edges, 8 waves per block
  const int etiles = E >> 4;
  const int eblocks = (etiles + 7) / 8;
  edge_kernel<<<eblocks, 256, E_SMEM_SZ, stream>>>(
      x, h, eidx, eattr, Wm1p, bm1, bm2, bc1, Wc2, bc2, agg, coord, E);

  // node pipeline + outputs
  const int ntiles = (N + 15) >> 4;
  const int nblocks = (ntiles + 7) / 8;
  node_kernel<<<nblocks, 256, N_SMEM_SZ, stream>>>(
      x, h, agg, coord, Wn1p, bn1, bn2, gamma, beta, out, N);
}
